// TokenMergerWithAttnHyperspace_55783035240573
// MI455X (gfx1250) — compile-verified, matrix operand order checked by reading, not run on hardware
//
#include <hip/hip_runtime.h>
#include <hip/hip_bf16.h>
#include <math.h>

// ---------------------------------------------------------------------------
// Problem constants (from reference)
// ---------------------------------------------------------------------------
constexpr int SEQ = 8192;   // S
constexpr int EMB = 2048;   // E
constexpr int NHEAD = 8;    // H
constexpr int HDIM = EMB / NHEAD;   // 256
constexpr int NGRP = 256;   // G
constexpr int NN = 4, NA = 4;
constexpr int ROWS = NGRP * NN;     // 1024 gathered rows per space
constexpr float ALPHA_C = 1.1f, BETA_C = 1.2f;
constexpr float MINN = 1e-15f;
constexpr float MAXNORM = 1.0f - 1e-5f;

typedef __attribute__((ext_vector_type(16))) _Float16 v16h;
typedef __attribute__((ext_vector_type(8)))  _Float16 v8h;
typedef __attribute__((ext_vector_type(8)))  float    v8f;

// ---------------------------------------------------------------------------
// Block-wide sum reduce (blockDim.x == 256, wave32)
// ---------------------------------------------------------------------------
__device__ __forceinline__ float block_reduce_sum(float v, float* sbuf) {
#pragma unroll
  for (int o = 16; o > 0; o >>= 1) v += __shfl_down(v, o, 32);
  __syncthreads();
  if ((threadIdx.x & 31) == 0) sbuf[threadIdx.x >> 5] = v;
  __syncthreads();
  float tot = 0.f;
#pragma unroll
  for (int i = 0; i < 8; ++i) tot += sbuf[i];
  return tot;
}

__device__ __forceinline__ float artanh_clip(float x) {
  float xc = fminf(fmaxf(x, -1.0f + 1e-5f), 1.0f - 1e-5f);
  return 0.5f * (log1pf(xc) - log1pf(-xc));
}

// exp_map scale factor for a raw vector with squared-norm ss:
// returns c such that exp_map(v) = c * v (tanh + project folded in)
__device__ __forceinline__ float expmap_scale(float ss) {
  float n = fmaxf(sqrtf(ss), MINN);
  float t = tanhf(fminf(n, 15.0f));
  float c = t / n;
  if (t > MAXNORM) c *= MAXNORM / t;
  return c;
}

// ---------------------------------------------------------------------------
// Kernel 1: hyper = mobius_add(exp_map(prompt), exp_map(sinusoidal_pe))
// ---------------------------------------------------------------------------
__global__ __launch_bounds__(256)
void build_hyper_kernel(const float* __restrict__ x, float* __restrict__ hyper) {
  __shared__ float sbuf[8];
  const int row = blockIdx.x;
  const int t = threadIdx.x;
  const float* xr = x + (size_t)row * EMB;

  float xv[8], pv[8];
  float sxx = 0.f, spp = 0.f, sxp = 0.f;
  const float kdiv = -9.210340371976184f / (float)EMB;  // -ln(10000)/E
#pragma unroll
  for (int j = 0; j < 8; ++j) {
    int e = t + 256 * j;
    float xx = xr[e];
    int p2 = (e >> 1) << 1;  // 2 * (e/2)
    float div = expf((float)p2 * kdiv);
    float ang = (float)row * div;
    float pe = (e & 1) ? cosf(ang) : sinf(ang);
    xv[j] = xx; pv[j] = pe;
    sxx += xx * xx; spp += pe * pe; sxp += xx * pe;
  }
  sxx = block_reduce_sum(sxx, sbuf);
  spp = block_reduce_sum(spp, sbuf);
  sxp = block_reduce_sum(sxp, sbuf);

  float cx = expmap_scale(sxx);
  float cp = expmap_scale(spp);
  float x2 = cx * cx * sxx;
  float y2 = cp * cp * spp;
  float xy = cx * cp * sxp;
  float den = fmaxf(1.0f + 2.0f * xy + x2 * y2, MINN);
  float ca = (1.0f + 2.0f * xy + y2) / den * cx;
  float cb = (1.0f - x2) / den * cp;

  float* hr = hyper + (size_t)row * EMB;
#pragma unroll
  for (int j = 0; j < 8; ++j) hr[t + 256 * j] = ca * xv[j] + cb * pv[j];
}

// ---------------------------------------------------------------------------
// Kernel 2: f32 -> f16 convert (weights)
// ---------------------------------------------------------------------------
__global__ __launch_bounds__(256)
void cvt_f16_kernel(const float* __restrict__ src, _Float16* __restrict__ dst, int n) {
  int i = blockIdx.x * 256 + threadIdx.x;
  if (i < n) dst[i] = (_Float16)src[i];
}

// ---------------------------------------------------------------------------
// Kernel 3: gather rows (by index) into contiguous f16 matrix
// ---------------------------------------------------------------------------
__global__ __launch_bounds__(256)
void gather_f16_kernel(const float* __restrict__ src, const int* __restrict__ idx,
                       _Float16* __restrict__ dst) {
  int row = blockIdx.x;
  int sr = idx[row];
  const float* s = src + (size_t)sr * EMB;
  _Float16* d = dst + (size_t)row * EMB;
#pragma unroll
  for (int j = 0; j < 8; ++j) {
    int e = threadIdx.x + 256 * j;
    d[e] = (_Float16)s[e];
  }
}

// ---------------------------------------------------------------------------
// Kernel 4: WMMA GEMM  C[M,N] = A[M,K] * B[N,K]^T + bias[N]
// A,B f16 row-major (K contiguous), C f32.
// 128x128 macro tile / 256 threads (8 waves, 4 Mw x 2 Nw, wave tile 32x64).
// K-step 64, double-buffered LDS filled with global_load_async_to_lds_b128
// (ASYNCcnt DMA), fragments read from LDS as ds_load_b128 pairs matching the
// CDNA5 WMMA register layout.
// ---------------------------------------------------------------------------
constexpr int KSTEP = 64;
constexpr int ROWH = KSTEP + 8;     // padded LDS row stride (halfs)

__global__ __launch_bounds__(256)
void wmma_gemm_nt(const _Float16* __restrict__ A, const _Float16* __restrict__ B,
                  const float* __restrict__ bias, float* __restrict__ C,
                  int M, int N, int K) {
  __shared__ _Float16 As[2][128 * ROWH];
  __shared__ _Float16 Bs[2][128 * ROWH];

  const int t = threadIdx.x;
  const int lane = t & 31;
  const int wave = t >> 5;
  const int wm = wave >> 1;          // 0..3
  const int wn = wave & 1;           // 0..1
  const int rih = lane & 15;         // row/col within 16
  const int kh  = (lane >> 4) << 3;  // 0 or 8 (K sub-offset per ISA layout)

  const int bm0 = blockIdx.y * 128;
  const int bn0 = blockIdx.x * 128;

  // Async DMA of one K-step tile pair into LDS buffer `buf`.
  // Tile = 128 rows x 64 halfs = 1024 chunks of 16B; 4 chunks per thread.
  auto issue_tile = [&](int buf, int k0) {
    unsigned lA0 = (unsigned)(uintptr_t)(&As[buf][0]);
    unsigned lB0 = (unsigned)(uintptr_t)(&Bs[buf][0]);
#pragma unroll
    for (int i = 0; i < 4; ++i) {
      int c = t + 256 * i;           // 0..1023
      int row = c >> 3;              // 64 halfs = 8 chunks per row
      int colh = (c & 7) * 8;
      unsigned lA = lA0 + (unsigned)(row * ROWH + colh) * 2u;
      unsigned lB = lB0 + (unsigned)(row * ROWH + colh) * 2u;
      unsigned long long gA =
          (unsigned long long)(uintptr_t)(A + (size_t)(bm0 + row) * K + k0 + colh);
      unsigned long long gB =
          (unsigned long long)(uintptr_t)(B + (size_t)(bn0 + row) * K + k0 + colh);
      asm volatile("global_load_async_to_lds_b128 %0, %1, off"
                   :: "v"(lA), "v"(gA) : "memory");
      asm volatile("global_load_async_to_lds_b128 %0, %1, off"
                   :: "v"(lB), "v"(gB) : "memory");
    }
  };

  v8f acc[2][4] = {};

  issue_tile(0, 0);
  asm volatile("s_wait_asynccnt 0" ::: "memory");
  __syncthreads();

  const int nk = K / KSTEP;
  for (int kt = 0; kt < nk; ++kt) {
    const int buf = kt & 1;
    if (kt + 1 < nk) issue_tile(buf ^ 1, (kt + 1) * KSTEP);

#pragma unroll
    for (int ks = 0; ks < KSTEP; ks += 32) {
      v16h af[2], bf[4];
#pragma unroll
      for (int i = 0; i < 2; ++i) {
        const _Float16* p = &As[buf][(wm * 32 + i * 16 + rih) * ROWH + ks + kh];
        v8h lo = *(const v8h*)p;          // K = ks+kh .. +7
        v8h hi = *(const v8h*)(p + 16);   // K = ks+kh+16 .. +23
        af[i] = __builtin_shufflevector(lo, hi, 0,1,2,3,4,5,6,7,8,9,10,11,12,13,14,15);
      }
#pragma unroll
      for (int j = 0; j < 4; ++j) {
        const _Float16* p = &Bs[buf][(wn * 64 + j * 16 + rih) * ROWH + ks + kh];
        v8h lo = *(const v8h*)p;
        v8h hi = *(const v8h*)(p + 16);
        bf[j] = __builtin_shufflevector(lo, hi, 0,1,2,3,4,5,6,7,8,9,10,11,12,13,14,15);
      }
#pragma unroll
      for (int i = 0; i < 2; ++i)
#pragma unroll
        for (int j = 0; j < 4; ++j)
          acc[i][j] = __builtin_amdgcn_wmma_f32_16x16x32_f16(
              false, af[i], false, bf[j], (short)0, acc[i][j], false, false);
    }

    // close double-buffer hazard: DMA landed + all waves done reading `buf`
    asm volatile("s_wait_asynccnt 0" ::: "memory");
    __syncthreads();
  }

  // Epilogue: lanes 0-15: VGPR r -> M=r, N=lane; lanes 16-31: M=8+r, N=lane-16
  const int mref = bm0 + wm * 32 + ((lane >> 4) << 3);
  const int nref = bn0 + wn * 64 + (lane & 15);
#pragma unroll
  for (int i = 0; i < 2; ++i) {
#pragma unroll
    for (int j = 0; j < 4; ++j) {
      int n = nref + j * 16;
      float bv = bias[n];
#pragma unroll
      for (int r = 0; r < 8; ++r) {
        int m = mref + i * 16 + r;
        C[(size_t)m * N + n] = acc[i][j][r] + bv;
      }
    }
  }
}

// ---------------------------------------------------------------------------
// Kernel 5: per-group multi-head attention (4 queries x 4 keys, 8 heads)
// ---------------------------------------------------------------------------
__global__ __launch_bounds__(256)
void attention_kernel(const float* __restrict__ Qe, const float* __restrict__ Ke,
                      const float* __restrict__ Ve, _Float16* __restrict__ Oe,
                      const float* __restrict__ Qh, const float* __restrict__ Kh,
                      const float* __restrict__ Vh, _Float16* __restrict__ Oh) {
  __shared__ float sc[NHEAD][NN][NA];
  const int g = blockIdx.x;
  const float* Q = blockIdx.y ? Qh : Qe;
  const float* Kp = blockIdx.y ? Kh : Ke;
  const float* V = blockIdx.y ? Vh : Ve;
  _Float16* O = blockIdx.y ? Oh : Oe;
  const int t = threadIdx.x;

  if (t < NHEAD * NN * NA) {
    int h = t >> 4, qi = (t >> 2) & 3, ki = t & 3;
    const float* q = Q + (size_t)(g * NN + qi) * EMB + h * HDIM;
    const float* k = Kp + (size_t)(g * NA + ki) * EMB + h * HDIM;
    float s = 0.f;
    for (int c = 0; c < HDIM; ++c) s += q[c] * k[c];
    sc[h][qi][ki] = s * 0.0625f;
  }
  __syncthreads();
  if (t < NHEAD * NN) {
    int h = t >> 2, qi = t & 3;
    float mx = sc[h][qi][0];
#pragma unroll
    for (int ki = 1; ki < NA; ++ki) mx = fmaxf(mx, sc[h][qi][ki]);
    float ex[NA], sum = 0.f;
#pragma unroll
    for (int ki = 0; ki < NA; ++ki) { ex[ki] = expf(sc[h][qi][ki] - mx); sum += ex[ki]; }
#pragma unroll
    for (int ki = 0; ki < NA; ++ki) sc[h][qi][ki] = ex[ki] / sum;
  }
  __syncthreads();
  for (int it = 0; it < 32; ++it) {
    int o = t + 256 * it;
    int qi = o >> 11;
    int col = o & (EMB - 1);
    int h = col >> 8;
    float a = 0.f;
#pragma unroll
    for (int ki = 0; ki < NA; ++ki)
      a += sc[h][qi][ki] * V[(size_t)(g * NA + ki) * EMB + col];
    O[(size_t)(g * NN + qi) * EMB + col] = (_Float16)a;
  }
}

// ---------------------------------------------------------------------------
// Kernel 6: per-group merge -> ms_h (mobius), ms_e (linear)
// ---------------------------------------------------------------------------
__global__ __launch_bounds__(256)
void group_merge_kernel(const int* __restrict__ noun_idx, const int* __restrict__ attr_idx,
                        const float* __restrict__ hyper, const float* __restrict__ prompt,
                        const float* __restrict__ ao_h, const float* __restrict__ ao_e,
                        float* __restrict__ ms_h, float* __restrict__ ms_e) {
  __shared__ float sbuf[8];
  const int g = blockIdx.x;
  const int t = threadIdx.x;

  float accA[8] = {0,0,0,0,0,0,0,0};
  float accB[8] = {0,0,0,0,0,0,0,0};
  float accE[8] = {0,0,0,0,0,0,0,0};

  for (int i = 0; i < NN; ++i) {
    int nr = noun_idx[g * NN + i];
    const float* nh = hyper + (size_t)nr * EMB;
    const float* ne = prompt + (size_t)nr * EMB;
    const float* oh = ao_h + (size_t)(g * NN + i) * EMB;
    const float* oe = ao_e + (size_t)(g * NN + i) * EMB;
    float a[8], b[8];
    float sxx = 0.f, syy = 0.f, sxy = 0.f;
#pragma unroll
    for (int j = 0; j < 8; ++j) {
      int e = t + 256 * j;
      float av = oh[e], bv = nh[e];
      a[j] = av; b[j] = bv;
      sxx += av * av; syy += bv * bv; sxy += av * bv;
      accE[j] += ALPHA_C * (oe[e] + ne[e]);
    }
    sxx = block_reduce_sum(sxx, sbuf);
    syy = block_reduce_sum(syy, sbuf);
    sxy = block_reduce_sum(sxy, sbuf);
    float den = fmaxf(1.0f + 2.0f * sxy + sxx * syy, MINN);
    float c1 = (1.0f + 2.0f * sxy + syy) / den;
    float c2 = (1.0f - sxx) / den;
#pragma unroll
    for (int j = 0; j < 8; ++j) accA[j] += ALPHA_C * (c1 * a[j] + c2 * b[j]);
  }

  for (int i = 0; i < NA; ++i) {
    int ar = attr_idx[g * NA + i];
    const float* ah = hyper + (size_t)ar * EMB;
    const float* ae = prompt + (size_t)ar * EMB;
#pragma unroll
    for (int j = 0; j < 8; ++j) {
      int e = t + 256 * j;
      accB[j] += BETA_C * ah[e];
      accE[j] += BETA_C * ae[e];
    }
  }

  float sxx = 0.f, syy = 0.f, sxy = 0.f;
#pragma unroll
  for (int j = 0; j < 8; ++j) {
    sxx += accA[j] * accA[j]; syy += accB[j] * accB[j]; sxy += accA[j] * accB[j];
  }
  sxx = block_reduce_sum(sxx, sbuf);
  syy = block_reduce_sum(syy, sbuf);
  sxy = block_reduce_sum(sxy, sbuf);
  float den = fmaxf(1.0f + 2.0f * sxy + sxx * syy, MINN);
  float c1 = (1.0f + 2.0f * sxy + syy) / den;
  float c2 = (1.0f - sxx) / den;
#pragma unroll
  for (int j = 0; j < 8; ++j) {
    int e = t + 256 * j;
    ms_h[(size_t)g * EMB + e] = c1 * accA[j] + c2 * accB[j];
    ms_e[(size_t)g * EMB + e] = accE[j];
  }
}

// ---------------------------------------------------------------------------
// Kernel 7: row-type map (-1 normal, -2 zeroed, >=0 group id of main row)
// ---------------------------------------------------------------------------
__global__ void map_init_kernel(int* __restrict__ map) {
  map[blockIdx.x * 256 + threadIdx.x] = -1;
}
__global__ void map_scatter_kernel(const int* __restrict__ noun_idx,
                                   const int* __restrict__ attr_idx,
                                   int* __restrict__ map) {
  int t = blockIdx.x * 256 + threadIdx.x;
  if (t >= NGRP * (NN + NA)) return;
  int g = t >> 3, j = t & 7;
  if (j < NN) {
    int r = noun_idx[g * NN + j];
    map[r] = (j == 0) ? g : -2;
  } else {
    map[attr_idx[g * NA + (j - NN)]] = -2;
  }
}

// ---------------------------------------------------------------------------
// Kernel 8: out = eu' + 0.1 * log_map(hyper')
// ---------------------------------------------------------------------------
__global__ __launch_bounds__(256)
void finalize_kernel(const int* __restrict__ map,
                     const float* __restrict__ ms_h, const float* __restrict__ ms_e,
                     const float* __restrict__ hyper, const float* __restrict__ prompt,
                     float* __restrict__ out) {
  __shared__ float sbuf[8];
  const int row = blockIdx.x;
  const int t = threadIdx.x;
  const int m = map[row];
  float* orow = out + (size_t)row * EMB;
  if (m == -2) {
#pragma unroll
    for (int j = 0; j < 8; ++j) orow[t + 256 * j] = 0.f;
    return;
  }
  const float* h = (m >= 0) ? ms_h + (size_t)m * EMB : hyper + (size_t)row * EMB;
  const float* e = (m >= 0) ? ms_e + (size_t)m * EMB : prompt + (size_t)row * EMB;
  float hv[8];
  float ss = 0.f;
#pragma unroll
  for (int j = 0; j < 8; ++j) {
    hv[j] = h[t + 256 * j];
    ss += hv[j] * hv[j];
  }
  ss = block_reduce_sum(ss, sbuf);
  float yn = fmaxf(sqrtf(ss), MINN);
  float c = artanh_clip(yn) / yn;
#pragma unroll
  for (int j = 0; j < 8; ++j) orow[t + 256 * j] = e[t + 256 * j] + 0.1f * c * hv[j];
}

// ---------------------------------------------------------------------------
// Host launch
// ---------------------------------------------------------------------------
extern "C" void kernel_launch(void* const* d_in, const int* in_sizes, int n_in,
                              void* d_out, int out_size, void* d_ws, size_t ws_size,
                              hipStream_t stream) {
  const float* prompt = (const float*)d_in[0];
  const float* w_in   = (const float*)d_in[1];
  const float* b_in   = (const float*)d_in[2];
  const float* w_out  = (const float*)d_in[3];
  const float* b_out  = (const float*)d_in[4];
  const int* noun_idx = (const int*)d_in[5];
  const int* attr_idx = (const int*)d_in[6];
  float* out = (float*)d_out;
  (void)in_sizes; (void)n_in; (void)out_size; (void)ws_size;

  char* p = (char*)d_ws;
  auto alloc = [&](size_t bytes) -> char* {
    char* r = p; p += (bytes + 255) & ~(size_t)255; return r;
  };
  float*    hyper = (float*)alloc((size_t)SEQ * EMB * sizeof(float));
  _Float16* w16   = (_Float16*)alloc((size_t)4 * EMB * EMB * sizeof(_Float16));
  _Float16* Ne    = (_Float16*)alloc((size_t)ROWS * EMB * sizeof(_Float16));
  _Float16* Ae    = (_Float16*)alloc((size_t)ROWS * EMB * sizeof(_Float16));
  _Float16* Nh    = (_Float16*)alloc((size_t)ROWS * EMB * sizeof(_Float16));
  _Float16* Ah    = (_Float16*)alloc((size_t)ROWS * EMB * sizeof(_Float16));
  float*    Qe    = (float*)alloc((size_t)ROWS * EMB * sizeof(float));
  float*    Ke    = (float*)alloc((size_t)ROWS * EMB * sizeof(float));
  float*    Ve    = (float*)alloc((size_t)ROWS * EMB * sizeof(float));
  float*    Qh    = (float*)alloc((size_t)ROWS * EMB * sizeof(float));
  float*    Kh    = (float*)alloc((size_t)ROWS * EMB * sizeof(float));
  float*    Vh    = (float*)alloc((size_t)ROWS * EMB * sizeof(float));
  float*    msh   = (float*)alloc((size_t)NGRP * EMB * sizeof(float));
  float*    mse   = (float*)alloc((size_t)NGRP * EMB * sizeof(float));
  int*      map   = (int*)alloc((size_t)SEQ * sizeof(int));
  _Float16* AOe16 = Ne;  // attention output f16 (euclid) reuses Ne
  _Float16* AOh16 = Nh;  // attention output f16 (hyper)  reuses Nh
  float*    AOe   = Qe;  // projected attention output reuses Qe
  float*    AOh   = Qh;

  build_hyper_kernel<<<SEQ, 256, 0, stream>>>(prompt, hyper);

  {
    int n_in_w = 3 * EMB * EMB;
    cvt_f16_kernel<<<(n_in_w + 255) / 256, 256, 0, stream>>>(w_in, w16, n_in_w);
    int n_out_w = EMB * EMB;
    cvt_f16_kernel<<<(n_out_w + 255) / 256, 256, 0, stream>>>(w_out, w16 + (size_t)3 * EMB * EMB, n_out_w);
  }

  gather_f16_kernel<<<ROWS, 256, 0, stream>>>(prompt, noun_idx, Ne);
  gather_f16_kernel<<<ROWS, 256, 0, stream>>>(prompt, attr_idx, Ae);
  gather_f16_kernel<<<ROWS, 256, 0, stream>>>(hyper,  noun_idx, Nh);
  gather_f16_kernel<<<ROWS, 256, 0, stream>>>(hyper,  attr_idx, Ah);

  dim3 ggrid(EMB / 128, ROWS / 128);   // (16, 8)
  const _Float16* Wq = w16;
  const _Float16* Wk = w16 + (size_t)EMB * EMB;
  const _Float16* Wv = w16 + (size_t)2 * EMB * EMB;
  const _Float16* Wo = w16 + (size_t)3 * EMB * EMB;
  wmma_gemm_nt<<<ggrid, 256, 0, stream>>>(Ne, Wq, b_in,            Qe, ROWS, EMB, EMB);
  wmma_gemm_nt<<<ggrid, 256, 0, stream>>>(Ae, Wk, b_in + EMB,      Ke, ROWS, EMB, EMB);
  wmma_gemm_nt<<<ggrid, 256, 0, stream>>>(Ae, Wv, b_in + 2 * EMB,  Ve, ROWS, EMB, EMB);
  wmma_gemm_nt<<<ggrid, 256, 0, stream>>>(Nh, Wq, b_in,            Qh, ROWS, EMB, EMB);
  wmma_gemm_nt<<<ggrid, 256, 0, stream>>>(Ah, Wk, b_in + EMB,      Kh, ROWS, EMB, EMB);
  wmma_gemm_nt<<<ggrid, 256, 0, stream>>>(Ah, Wv, b_in + 2 * EMB,  Vh, ROWS, EMB, EMB);

  attention_kernel<<<dim3(NGRP, 2), 256, 0, stream>>>(Qe, Ke, Ve, AOe16,
                                                      Qh, Kh, Vh, AOh16);

  wmma_gemm_nt<<<ggrid, 256, 0, stream>>>(AOe16, Wo, b_out, AOe, ROWS, EMB, EMB);
  wmma_gemm_nt<<<ggrid, 256, 0, stream>>>(AOh16, Wo, b_out, AOh, ROWS, EMB, EMB);

  map_init_kernel<<<SEQ / 256, 256, 0, stream>>>(map);
  map_scatter_kernel<<<(NGRP * (NN + NA) + 255) / 256, 256, 0, stream>>>(noun_idx, attr_idx, map);

  group_merge_kernel<<<NGRP, 256, 0, stream>>>(noun_idx, attr_idx, hyper, prompt,
                                               AOh, AOe, msh, mse);

  finalize_kernel<<<SEQ, 256, 0, stream>>>(map, msh, mse, hyper, prompt, out);
}
